// AttentionCellEncoder_9466107920688
// MI455X (gfx1250) — compile-verified, matrix operand order checked by reading, not run on hardware
//
#include <hip/hip_runtime.h>

typedef unsigned short u16;
typedef unsigned int   u32;

typedef __attribute__((ext_vector_type(16))) __bf16 v16bf;
typedef __attribute__((ext_vector_type(8)))  float  v8f;
typedef __attribute__((ext_vector_type(4)))  u32    u32x4;
typedef __attribute__((ext_vector_type(2)))  u32    u32x2;
typedef __attribute__((ext_vector_type(4)))  float  f32x4;

#define NUM_CELLS_  2048
#define MAX_LEN_    64
#define IN_DIM_     768
#define HID_        512
#define NH_         8
#define HD_         64
#define OUT_DIM_    256

// ---- LDS layout (u16 units) ----
// Region 0 (96KB): gathered bf16 activations x[64][768]; after projections it is
// dead and re-used for per-wave softmax P tiles and the bf16 context matrix.
#define LDS_X    0            // 64*768            = 49152 u16
#define LDS_P    0            // 8 waves * 16*64   =  8192 u16 (aliases X)
#define LDS_CTX  8192         // 64*512            = 32768 u16 (aliases X)
#define LDS_Q    49152        // 64*512
#define LDS_K    81920        // 64*512
#define LDS_V    114688       // transposed: [512][64]
#define LDS_U16_TOTAL 147456
#define LDS_F32_BYTE_OFF 294912   // 1024 floats: pooled[512] + t[512]
#define LDS_BYTES (294912 + 4096)

__device__ __forceinline__ u16 f2bf(float f) {
  u32 x = __float_as_uint(f);
  x += 0x7fffu + ((x >> 16) & 1u);          // round-to-nearest-even
  return (u16)(x >> 16);
}
__device__ __forceinline__ float bf2f(u16 h) {
  return __uint_as_float(((u32)h) << 16);
}

union V16U { u32x4 q[2]; v16bf v; };

// Load one WMMA 16-bit operand (A or B) for the CDNA5 16x32 layout from a
// row-major [R][ldk] bf16 array. Caller passes p = &base[(row)*ldk + k0].
// lane<16  (khalf=0): K = k0+0..7  and k0+16..23
// lane>=16 (khalf=8): K = k0+8..15 and k0+24..31
__device__ __forceinline__ v16bf ld16(const u16* __restrict__ p, int khalf) {
  V16U t;
  t.q[0] = *(const u32x4*)(p + khalf);
  t.q[1] = *(const u32x4*)(p + khalf + 16);
  return t.v;
}

__device__ __forceinline__ v8f wmma_bf16(v16bf a, v16bf b, v8f c) {
  return __builtin_amdgcn_wmma_f32_16x16x32_bf16(
      /*neg_a=*/false, a, /*neg_b=*/false, b,
      /*c_mod=*/(short)0, c, /*reuse_a=*/false, /*reuse_b=*/false);
}

// Store a 16x16 f32 C-tile (+bias) from the CDNA5 C layout:
// element j of the v8f on lane L is (M = j + 8*(L>>4), N = L&15) within tile.
__device__ __forceinline__ void store_proj_tile(u16* __restrict__ sm, int sel, int mt,
                                                int ncol, float bias, v8f acc, int lane) {
  const int mbase = mt * 16 + ((lane & 16) ? 8 : 0);
#pragma unroll
  for (int j = 0; j < 8; ++j) {
    const float v = acc[j] + bias;
    const int m = mbase + j;
    if (sel == 0)      sm[LDS_Q + m * HID_ + ncol] = f2bf(v);
    else if (sel == 1) sm[LDS_K + m * HID_ + ncol] = f2bf(v);
    else               sm[LDS_V + ncol * 64 + m]   = f2bf(v);   // transposed V
  }
}

// -------------------------------------------------------------------------
// Kernel 1: fold MHA in_proj into the q/k/v projections.
//   Wc[sel][n][d] = sum_h W_in[sel*512+n][h] * Wsel[h][d]   (bf16, to ws)
//   bc[sel*512+n] = b_in[sel*512+n] + sum_h W_in[..][h]*bsel[h]
// -------------------------------------------------------------------------
__global__ __launch_bounds__(256)
void combine_weights_kernel(const float* __restrict__ W_in, const float* __restrict__ b_in,
                            const float* __restrict__ Wq, const float* __restrict__ bq,
                            const float* __restrict__ Wk, const float* __restrict__ bk,
                            const float* __restrict__ Wv, const float* __restrict__ bv,
                            u16* __restrict__ Wc, float* __restrict__ bc) {
  const long i = (long)blockIdx.x * 256 + threadIdx.x;
  const long NW = (long)3 * HID_ * IN_DIM_;    // 1536*768
  if (i < NW) {
    const int n = (int)(i / IN_DIM_);          // 0..1535
    const int d = (int)(i - (long)n * IN_DIM_);
    const int sel = n >> 9;
    const float* Wsrc = (sel == 0) ? Wq : ((sel == 1) ? Wk : Wv);
    const float* wi = W_in + (long)n * HID_;
    float acc = 0.f;
    for (int h = 0; h < HID_; ++h) acc += wi[h] * Wsrc[h * IN_DIM_ + d];
    Wc[i] = f2bf(acc);
  } else if (i < NW + 3 * HID_) {
    const int n = (int)(i - NW);
    const int sel = n >> 9;
    const float* bsrc = (sel == 0) ? bq : ((sel == 1) ? bk : bv);
    const float* wi = W_in + (long)n * HID_;
    float acc = b_in[n];
    for (int h = 0; h < HID_; ++h) acc += wi[h] * bsrc[h];
    bc[n] = acc;
  }
}

// -------------------------------------------------------------------------
// Kernel 2: one workgroup (8 wave32) per cell. Fully fused:
// gather->bf16 LDS, WMMA qkv projection (with folded in_proj), per-head WMMA
// attention + masked softmax, WMMA P@V, masked mean pool, Wo+Wout mat-vecs.
// -------------------------------------------------------------------------
__global__ __launch_bounds__(256)
void cell_attn_kernel(const float* __restrict__ feat,
                      const u16* __restrict__ Wc, const float* __restrict__ bc,
                      const float* __restrict__ Wo, const float* __restrict__ bo,
                      const float* __restrict__ Wout, const float* __restrict__ bout,
                      const int* __restrict__ cidx, const int* __restrict__ clen,
                      float* __restrict__ out) {
  extern __shared__ char raw[];
  u16*   sm    = (u16*)raw;
  float* poolf = (float*)(raw + LDS_F32_BYTE_OFF);   // pooled[512], t[512]

  const int tid  = threadIdx.x;
  const int lane = tid & 31;
  const int wave = tid >> 5;
  const int lr   = lane & 15;
  const int khalf = (lane & 16) ? 8 : 0;
  const int cell = blockIdx.x;

  int len = clen[cell];
  if (len < 1) len = 1;
  if (len > MAX_LEN_) len = MAX_LEN_;

  // ---- phase 0: cell indices (alias the float scratch; dead until phase 4)
  int* idxs = (int*)poolf;
  if (tid < MAX_LEN_) idxs[tid] = cidx[cell * MAX_LEN_ + tid];
  __syncthreads();

  // ---- phase 1: gather + f32->bf16 into LDS  (64 rows x 768)
  for (int i = tid; i < MAX_LEN_ * (IN_DIM_ / 4); i += 256) {
    const int row = i / (IN_DIM_ / 4);
    const int c4  = (i - row * (IN_DIM_ / 4)) * 4;
    const f32x4 f = *(const f32x4*)(feat + (size_t)idxs[row] * IN_DIM_ + c4);
    u32x2 pk;
    pk.x = (u32)f2bf(f.x) | ((u32)f2bf(f.y) << 16);
    pk.y = (u32)f2bf(f.z) | ((u32)f2bf(f.w) << 16);
    *(u32x2*)(sm + LDS_X + row * IN_DIM_ + c4) = pk;
  }
  __syncthreads();

  // ---- phase 2: q/k/v = x @ Wc^T + bc   (M=64, K=768, N=3*512)
  // 96 N-tiles of 16 over 8 waves; B tile reused across the 4 M-tiles.
  for (int t = 0; t < 12; ++t) {
    const int ntg  = wave + (t << 3);          // 0..95
    const int sel  = ntg >> 5;                 // 0=q,1=k,2=v
    const int ntl  = ntg & 31;
    const int ncol = (ntl << 4) | lr;
    const u16* wrow = Wc + (size_t)(sel * HID_ + ncol) * IN_DIM_;
    __builtin_prefetch(wrow, 0, 1);
    __builtin_prefetch(wrow + IN_DIM_ / 2, 0, 1);
    const float bias = bc[sel * HID_ + ncol];
    v8f a0 = {}, a1 = {}, a2 = {}, a3 = {};
    for (int kt = 0; kt < IN_DIM_ / 32; ++kt) {
      const int k0 = kt * 32;
      const v16bf b  = ld16(wrow + k0, khalf);
      const v16bf x0 = ld16(sm + LDS_X + (0 * 16 + lr) * IN_DIM_ + k0, khalf);
      const v16bf x1 = ld16(sm + LDS_X + (1 * 16 + lr) * IN_DIM_ + k0, khalf);
      const v16bf x2 = ld16(sm + LDS_X + (2 * 16 + lr) * IN_DIM_ + k0, khalf);
      const v16bf x3 = ld16(sm + LDS_X + (3 * 16 + lr) * IN_DIM_ + k0, khalf);
      a0 = wmma_bf16(x0, b, a0);
      a1 = wmma_bf16(x1, b, a1);
      a2 = wmma_bf16(x2, b, a2);
      a3 = wmma_bf16(x3, b, a3);
    }
    store_proj_tile(sm, sel, 0, ncol, bias, a0, lane);
    store_proj_tile(sm, sel, 1, ncol, bias, a1, lane);
    store_proj_tile(sm, sel, 2, ncol, bias, a2, lane);
    store_proj_tile(sm, sel, 3, ncol, bias, a3, lane);
  }
  __syncthreads();   // also fences re-use of the X region by phase 3

  // ---- phase 3: attention. 32 (head, row-tile) pairs over 8 waves.
  u16* Pw = sm + LDS_P + wave * (16 * 64);
  const float NEG = -3.0e38f;
  for (int t = 0; t < 4; ++t) {
    const int pi = wave + (t << 3);            // 0..31
    const int h  = pi >> 2;
    const int rt = pi & 3;

    // scores S[16 x 64] = q_tile @ k^T  (K = head_dim = 64)
    v8f s0 = {}, s1 = {}, s2 = {}, s3 = {};
    for (int kt = 0; kt < 2; ++kt) {
      const int k0 = h * HD_ + kt * 32;
      const v16bf a  = ld16(sm + LDS_Q + (rt * 16 + lr) * HID_ + k0, khalf);
      const v16bf b0 = ld16(sm + LDS_K + (0 * 16 + lr) * HID_ + k0, khalf);
      const v16bf b1 = ld16(sm + LDS_K + (1 * 16 + lr) * HID_ + k0, khalf);
      const v16bf b2 = ld16(sm + LDS_K + (2 * 16 + lr) * HID_ + k0, khalf);
      const v16bf b3 = ld16(sm + LDS_K + (3 * 16 + lr) * HID_ + k0, khalf);
      s0 = wmma_bf16(a, b0, s0);
      s1 = wmma_bf16(a, b1, s1);
      s2 = wmma_bf16(a, b2, s2);
      s3 = wmma_bf16(a, b3, s3);
    }

    // masked softmax over the 64 key columns; columns live in lane%16 x 4 tiles
#pragma unroll
    for (int j = 0; j < 8; ++j) {
      float v0 = s0[j] * 0.125f; if ( 0 + lr >= len) v0 = NEG;
      float v1 = s1[j] * 0.125f; if (16 + lr >= len) v1 = NEG;
      float v2 = s2[j] * 0.125f; if (32 + lr >= len) v2 = NEG;
      float v3 = s3[j] * 0.125f; if (48 + lr >= len) v3 = NEG;
      float m = fmaxf(fmaxf(v0, v1), fmaxf(v2, v3));
      m = fmaxf(m, __shfl_xor(m, 1));
      m = fmaxf(m, __shfl_xor(m, 2));
      m = fmaxf(m, __shfl_xor(m, 4));
      m = fmaxf(m, __shfl_xor(m, 8));      // row groups are 16-lane halves
      v0 = __expf(v0 - m); v1 = __expf(v1 - m);
      v2 = __expf(v2 - m); v3 = __expf(v3 - m);
      float ss = v0 + v1 + v2 + v3;
      ss += __shfl_xor(ss, 1);
      ss += __shfl_xor(ss, 2);
      ss += __shfl_xor(ss, 4);
      ss += __shfl_xor(ss, 8);
      const float inv = 1.0f / ss;
      const int ml = j + ((lane & 16) ? 8 : 0);
      u16* pr = Pw + ml * 64 + lr;
      pr[0]  = f2bf(v0 * inv);
      pr[16] = f2bf(v1 * inv);
      pr[32] = f2bf(v2 * inv);
      pr[48] = f2bf(v3 * inv);
    }

    // ctx[16 x 64] = P @ V   (B from transposed V: row = head col, contiguous K)
    v8f c0 = {}, c1 = {}, c2 = {}, c3 = {};
    for (int kt = 0; kt < 2; ++kt) {
      const int k0 = kt * 32;
      const v16bf a  = ld16(Pw + lr * 64 + k0, khalf);
      const v16bf b0 = ld16(sm + LDS_V + (h * HD_ +  0 + lr) * 64 + k0, khalf);
      const v16bf b1 = ld16(sm + LDS_V + (h * HD_ + 16 + lr) * 64 + k0, khalf);
      const v16bf b2 = ld16(sm + LDS_V + (h * HD_ + 32 + lr) * 64 + k0, khalf);
      const v16bf b3 = ld16(sm + LDS_V + (h * HD_ + 48 + lr) * 64 + k0, khalf);
      c0 = wmma_bf16(a, b0, c0);
      c1 = wmma_bf16(a, b1, c1);
      c2 = wmma_bf16(a, b2, c2);
      c3 = wmma_bf16(a, b3, c3);
    }
#pragma unroll
    for (int j = 0; j < 8; ++j) {
      const int m = rt * 16 + j + ((lane & 16) ? 8 : 0);
      u16* cr = sm + LDS_CTX + m * HID_ + h * HD_ + lr;
      cr[0]  = f2bf(c0[j]);
      cr[16] = f2bf(c1[j]);
      cr[32] = f2bf(c2[j]);
      cr[48] = f2bf(c3[j]);
    }
  }
  __syncthreads();

  // ---- phase 4: masked mean pool over valid rows (out_proj commutes w/ pool)
  const float invlen = 1.0f / (float)len;
  for (int n = tid; n < HID_; n += 256) {
    float a = 0.f;
    for (int l = 0; l < len; ++l) a += bf2f(sm[LDS_CTX + l * HID_ + n]);
    poolf[n] = a * invlen;
  }
  __syncthreads();

  // t = pooled @ Wo^T + bo
  float* tb = poolf + HID_;
  for (int n = tid; n < HID_; n += 256) {
    const float* wr = Wo + (size_t)n * HID_;
    float a = bo[n];
    for (int d = 0; d < HID_; ++d) a += poolf[d] * wr[d];
    tb[n] = a;
  }
  __syncthreads();

  // y = t @ Wout^T + bout
  if (tid < OUT_DIM_) {
    const float* wr = Wout + (size_t)tid * HID_;
    float a = bout[tid];
    for (int d = 0; d < HID_; ++d) a += tb[d] * wr[d];
    out[(size_t)cell * OUT_DIM_ + tid] = a;
  }
}

// -------------------------------------------------------------------------
extern "C" void kernel_launch(void* const* d_in, const int* in_sizes, int n_in,
                              void* d_out, int out_size, void* d_ws, size_t ws_size,
                              hipStream_t stream) {
  const float* feat = (const float*)d_in[0];
  const float* Wq   = (const float*)d_in[1];
  const float* bq   = (const float*)d_in[2];
  const float* Wk   = (const float*)d_in[3];
  const float* bk   = (const float*)d_in[4];
  const float* Wv   = (const float*)d_in[5];
  const float* bv   = (const float*)d_in[6];
  const float* W_in = (const float*)d_in[7];
  const float* b_in = (const float*)d_in[8];
  const float* Wo   = (const float*)d_in[9];
  const float* bo   = (const float*)d_in[10];
  const float* Wout = (const float*)d_in[11];
  const float* bout = (const float*)d_in[12];
  const int*   cidx = (const int*)d_in[13];
  const int*   clen = (const int*)d_in[14];
  float* out = (float*)d_out;

  // workspace: combined bf16 weights [3*512][768] + combined bias [1536] f32
  u16*   Wc = (u16*)d_ws;
  float* bc = (float*)((char*)d_ws + (size_t)3 * HID_ * IN_DIM_ * sizeof(u16));

  const long nco = (long)3 * HID_ * IN_DIM_ + 3 * HID_;
  combine_weights_kernel<<<(unsigned)((nco + 255) / 256), 256, 0, stream>>>(
      W_in, b_in, Wq, bq, Wk, bk, Wv, bv, Wc, bc);

  (void)hipFuncSetAttribute((const void*)cell_attn_kernel,
                            hipFuncAttributeMaxDynamicSharedMemorySize, LDS_BYTES);
  cell_attn_kernel<<<NUM_CELLS_, 256, LDS_BYTES, stream>>>(
      feat, Wc, bc, Wo, bo, Wout, bout, cidx, clen, out);
}
